// Arch18GraphEncoder_9345848836681
// MI455X (gfx1250) — matrix operand order repaired and checked by reading, not compiled, not run on hardware
//
#include <hip/hip_runtime.h>
#include <hip/hip_fp16.h>
#include <cstddef>
#include <cstdint>

// ---------------------------------------------------------------------------
// Arch18 graph encoder for MI455X (gfx1250, wave32, WMMA).
// All dense GEMMs run on v_wmma_f32_16x16x32_f16 (f16 in, f32 accumulate).
// ---------------------------------------------------------------------------

typedef __attribute__((ext_vector_type(16))) _Float16 v16h;
typedef __attribute__((ext_vector_type(8)))  _Float16 v8h;
typedef __attribute__((ext_vector_type(4)))  _Float16 v4h;
typedef __attribute__((ext_vector_type(8)))  float    v8f;

#define KSUB   16
#define MTOK   8
#define NSUB   16384
#define NFN    (NSUB * KSUB)     // 262144 nodes
#define NTOK   (NSUB / MTOK)     // 2048
#define NE     524288
#define HD     128
#define NSTEP  16
#define NHEAD  4
#define DHEAD  32
#define FFD    512
#define NB     64

__device__ __forceinline__ float lp_masked(float x) {
  return __builtin_isfinite(x) ? x : 0.0f;
}

// ---------------------------------------------------------------------------
// RWSE: build per-subgraph 16x16 transition matrices
// ---------------------------------------------------------------------------
__global__ void k_build_T(const int* __restrict__ src, const int* __restrict__ dst,
                          float* __restrict__ T) {
  int e = blockIdx.x * blockDim.x + threadIdx.x;
  if (e >= NE) return;
  int s = src[e], d = dst[e];
  int sub = s >> 4;
  T[(size_t)sub * 256 + (s & 15) * 16 + (d & 15)] = 1.0f;  // duplicates all write 1.0
}

__global__ void k_norm_T(float* __restrict__ T) {
  int r = blockIdx.x * blockDim.x + threadIdx.x;      // S*16 rows
  if (r >= NSUB * 16) return;
  float* row = T + (size_t)r * 16;
  float s = 0.f;
#pragma unroll
  for (int j = 0; j < 16; ++j) s += row[j];
  float inv = 1.0f / fmaxf(s, 1.0f);
#pragma unroll
  for (int j = 0; j < 16; ++j) row[j] *= inv;
}

// one thread per (subgraph,row); T tiles staged in LDS (16 subs per block)
__global__ void __launch_bounds__(256) k_rwse(const float* __restrict__ T,
                                              float* __restrict__ rw) {
  __shared__ float Tl[16 * 256];
  int ls = threadIdx.x >> 4;
  int i  = threadIdx.x & 15;
  int sub = blockIdx.x * 16 + ls;
#pragma unroll
  for (int j = 0; j < 16; ++j)
    Tl[ls * 256 + i * 16 + j] = T[(size_t)sub * 256 + i * 16 + j];
  __syncthreads();
  float row[16];
#pragma unroll
  for (int j = 0; j < 16; ++j) row[j] = (j == i) ? 1.0f : 0.0f;
  const float* Ts = &Tl[ls * 256];
  for (int step = 0; step < NSTEP; ++step) {
    float nr[16];
#pragma unroll
    for (int j = 0; j < 16; ++j) nr[j] = 0.0f;
#pragma unroll
    for (int k2 = 0; k2 < 16; ++k2) {
      float pv = row[k2];
#pragma unroll
      for (int j = 0; j < 16; ++j) nr[j] += pv * Ts[k2 * 16 + j];
    }
#pragma unroll
    for (int j = 0; j < 16; ++j) row[j] = nr[j];
    rw[((size_t)(sub * 16 + i)) * 16 + step] = row[i];
  }
}

// ---------------------------------------------------------------------------
// h = (atom_emb[x] + relu(rw@rwse_W + b) + relu(lp*logp_W + b)) * valid
// ---------------------------------------------------------------------------
__global__ void k_init_h(const int* __restrict__ x_ids, const unsigned char* __restrict__ validm,
                         const float* __restrict__ log_probs, const float* __restrict__ atom_emb,
                         const float* __restrict__ rw, const float* __restrict__ rwse_W,
                         const float* __restrict__ rwse_b, const float* __restrict__ logp_W,
                         const float* __restrict__ logp_b, float* __restrict__ h) {
  int idx = blockIdx.x * blockDim.x + threadIdx.x;   // NFN*HD
  int node = idx >> 7, c = idx & 127;
  float acc = rwse_b[c];
  const float* rr = rw + (size_t)node * 16;
#pragma unroll
  for (int s = 0; s < 16; ++s) acc += rr[s] * rwse_W[s * HD + c];
  float pe1 = fmaxf(acc, 0.0f);
  float lpv = lp_masked(log_probs[node >> 4]);
  float pe2 = fmaxf(lpv * logp_W[c] + logp_b[c], 0.0f);
  float xf = atom_emb[(size_t)x_ids[node] * HD + c];
  float vf = validm[node] ? 1.0f : 0.0f;
  h[(size_t)idx] = (xf + pe1 + pe2) * vf;
}

// ---------------------------------------------------------------------------
// GINE message scatter: agg[dst] += relu(h[src] + bond_emb[ea-1])
// ---------------------------------------------------------------------------
__global__ void k_gine_scatter(const int* __restrict__ src, const int* __restrict__ dst,
                               const int* __restrict__ ea_ids, const float* __restrict__ bond_emb,
                               const float* __restrict__ h, float* __restrict__ agg) {
  int t = blockIdx.x * blockDim.x + threadIdx.x;   // NE*32
  int e = t >> 5, cb = (t & 31) << 2;
  int s = src[e], d = dst[e], bid = ea_ids[e] - 1;
  const float4 hv = *(const float4*)(h + (size_t)s * HD + cb);
  const float4 bv = *(const float4*)(bond_emb + (size_t)bid * HD + cb);
  float* ag = agg + (size_t)d * HD + cb;
  unsafeAtomicAdd(ag + 0, fmaxf(hv.x + bv.x, 0.0f));
  unsafeAtomicAdd(ag + 1, fmaxf(hv.y + bv.y, 0.0f));
  unsafeAtomicAdd(ag + 2, fmaxf(hv.z + bv.z, 0.0f));
  unsafeAtomicAdd(ag + 3, fmaxf(hv.w + bv.w, 0.0f));
}

// u16 = (1+eps)*h + agg   (f16 for the WMMA GEMM)
__global__ void k_gine_fuse(const float* __restrict__ h, const float* __restrict__ agg,
                            const float* __restrict__ eps_p, _Float16* __restrict__ u16) {
  int i = blockIdx.x * blockDim.x + threadIdx.x;   // NFN*HD/4
  float e = 1.0f + *eps_p;
  const float4 hv = *(const float4*)(h + (size_t)i * 4);
  const float4 av = *(const float4*)(agg + (size_t)i * 4);
  v4h o;
  o[0] = (_Float16)(e * hv.x + av.x);
  o[1] = (_Float16)(e * hv.y + av.y);
  o[2] = (_Float16)(e * hv.z + av.z);
  o[3] = (_Float16)(e * hv.w + av.w);
  *(v4h*)(u16 + (size_t)i * 4) = o;
}

// ---------------------------------------------------------------------------
// Generic WMMA GEMM: out[rows x Nd] = act(A[rows x Kd] @ W[Kd x Nd] + bias)
// W pre-transposed+f16: Wt[n*Kd + k].  One 16x16 output tile per wave.
// ---------------------------------------------------------------------------
template <bool RELU, bool RES, bool VALID, bool OUT16>
__global__ void __launch_bounds__(256)
k_gemm_wmma(const _Float16* __restrict__ A, const _Float16* __restrict__ Wt,
            const float* __restrict__ bias, const float* __restrict__ res,
            const unsigned char* __restrict__ validm,
            float* __restrict__ outf, _Float16* __restrict__ outh,
            int Kd, int Nd) {
  const int lane  = threadIdx.x & 31;
  const int wave  = blockIdx.x * (blockDim.x >> 5) + (threadIdx.x >> 5);
  const int nT    = Nd >> 4;
  const int colB  = (wave % nT) << 4;
  const int rowB  = (wave / nT) << 4;
  const int half_ = lane >> 4;
  const int n     = lane & 15;          // A row (M) and B/D column (N)

  const _Float16* arow = A  + (size_t)(rowB + n) * Kd;
  const _Float16* brow = Wt + (size_t)(colB + n) * Kd;
  const float bv = bias ? bias[colB + n] : 0.0f;
  v8f acc = {bv, bv, bv, bv, bv, bv, bv, bv};

  const int aoff0 = half_ * 8;
  const int aoff1 = 16 + half_ * 8;
  const int boff  = half_ * 8;
  for (int kb = 0; kb < Kd; kb += 32) {
    v8h alo = *(const v8h*)(arow + kb + aoff0);
    v8h ahi = *(const v8h*)(arow + kb + aoff1);
    v8h blo = *(const v8h*)(brow + kb + boff);
    v8h bhi = *(const v8h*)(brow + kb + boff + 16);
    v16h av = __builtin_shufflevector(alo, ahi, 0,1,2,3,4,5,6,7,8,9,10,11,12,13,14,15);
    v16h bvv = __builtin_shufflevector(blo, bhi, 0,1,2,3,4,5,6,7,8,9,10,11,12,13,14,15);
    acc = __builtin_amdgcn_wmma_f32_16x16x32_f16(false, av, false, bvv, (short)0, acc,
                                                 false, false);
  }

  const int col = colB + n;
#pragma unroll
  for (int v = 0; v < 8; ++v) {
    int row = rowB + v + half_ * 8;
    float val = acc[v];
    if (RELU) val = fmaxf(val, 0.0f);
    size_t o = (size_t)row * Nd + col;
    if (RES) val += res[o];
    if (VALID) val *= (validm[row] ? 1.0f : 0.0f);
    if (OUT16) outh[o] = (_Float16)val;
    else       outf[o] = val;
  }
}

// transpose + convert f32 W[Kd x Nd] -> f16 Wt[Nd x Kd]
__global__ void k_convert_wt(const float* __restrict__ W, _Float16* __restrict__ Wt,
                             int Kd, int Nd) {
  int t = blockIdx.x * blockDim.x + threadIdx.x;
  if (t >= Kd * Nd) return;
  int nn = t / Kd, k = t % Kd;
  Wt[t] = (_Float16)W[(size_t)k * Nd + nn];
}

// ---------------------------------------------------------------------------
// LayerNorm over H=128: one wave per row
// ---------------------------------------------------------------------------
template <bool OUT16>
__global__ void __launch_bounds__(256)
k_ln(const float* __restrict__ x, const float* __restrict__ g, const float* __restrict__ b,
     float* __restrict__ outf, _Float16* __restrict__ outh, int rows) {
  int wave = blockIdx.x * (blockDim.x >> 5) + (threadIdx.x >> 5);
  int lane = threadIdx.x & 31;
  if (wave >= rows) return;
  const float* xr = x + (size_t)wave * HD;
  float4 vv = *(const float4*)(xr + lane * 4);
  float s  = vv.x + vv.y + vv.z + vv.w;
  float s2 = vv.x * vv.x + vv.y * vv.y + vv.z * vv.z + vv.w * vv.w;
#pragma unroll
  for (int d = 16; d > 0; d >>= 1) {
    s  += __shfl_xor(s, d, 32);
    s2 += __shfl_xor(s2, d, 32);
  }
  float mu  = s * (1.0f / HD);
  float var = s2 * (1.0f / HD) - mu * mu;
  float inv = rsqrtf(var + 1e-5f);
  int c = lane * 4;
  float xv[4] = {vv.x, vv.y, vv.z, vv.w};
#pragma unroll
  for (int j = 0; j < 4; ++j) {
    float val = (xv[j] - mu) * inv * g[c + j] + b[c + j];
    if (OUT16) outh[(size_t)wave * HD + c + j] = (_Float16)val;
    else       outf[(size_t)wave * HD + c + j] = val;
  }
}

__global__ void k_copy_roots(const float* __restrict__ h, float* __restrict__ x) {
  int idx = blockIdx.x * blockDim.x + threadIdx.x;    // NTOK*MTOK*HD
  int s = idx >> 7, c = idx & 127;
  x[(size_t)idx] = h[(size_t)s * 16 * HD + c];        // root node = s*K
}

// ---------------------------------------------------------------------------
// Attention over M=8 tokens, HEADS=4, DH=32, with log-prob bias.
// ---------------------------------------------------------------------------
__global__ void __launch_bounds__(256)
k_attn(const float* __restrict__ q, const float* __restrict__ kk, const float* __restrict__ vv,
       const float* __restrict__ log_probs, _Float16* __restrict__ o16) {
  int t = blockIdx.x * blockDim.x + threadIdx.x;      // NTOK*32
  int i = t & 7, head = (t >> 3) & 3, nt = t >> 5;
  const float scale = 0.17677669529663687f;           // 1/sqrt(32)
  size_t qoff = ((size_t)(nt * 8 + i)) * HD + head * DHEAD;
  float qv[32];
#pragma unroll
  for (int c = 0; c < 32; ++c) qv[c] = q[qoff + c];
  float sc[8];
#pragma unroll
  for (int j = 0; j < 8; ++j) {
    size_t ko = ((size_t)(nt * 8 + j)) * HD + head * DHEAD;
    float d = 0.0f;
#pragma unroll
    for (int c = 0; c < 32; ++c) d += qv[c] * kk[ko + c];
    sc[j] = d * scale + lp_masked(log_probs[nt * 8 + j]);
  }
  float mx = sc[0];
#pragma unroll
  for (int j = 1; j < 8; ++j) mx = fmaxf(mx, sc[j]);
  float se = 0.0f;
#pragma unroll
  for (int j = 0; j < 8; ++j) { sc[j] = expf(sc[j] - mx); se += sc[j]; }
  float inv = 1.0f / se;
  float ov[32];
#pragma unroll
  for (int c = 0; c < 32; ++c) ov[c] = 0.0f;
#pragma unroll
  for (int j = 0; j < 8; ++j) {
    float a = sc[j] * inv;
    size_t vo = ((size_t)(nt * 8 + j)) * HD + head * DHEAD;
#pragma unroll
    for (int c = 0; c < 32; ++c) ov[c] += a * vv[vo + c];
  }
#pragma unroll
  for (int c = 0; c < 32; ++c) o16[qoff + c] = (_Float16)ov[c];
}

// final: out[batch[nt]] += sum_m softmax(lp2)[m] * xln[nt,m,:]
__global__ void k_final(const float* __restrict__ xln, const float* __restrict__ log_probs,
                        const int* __restrict__ batch, float* __restrict__ out) {
  int t = blockIdx.x * blockDim.x + threadIdx.x;      // NTOK*HD
  int c = t & 127, nt = t >> 7;
  float l[8];
  float mx = -3.4e38f;
#pragma unroll
  for (int m = 0; m < 8; ++m) { l[m] = lp_masked(log_probs[nt * 8 + m]); mx = fmaxf(mx, l[m]); }
  float se = 0.0f;
#pragma unroll
  for (int m = 0; m < 8; ++m) { l[m] = expf(l[m] - mx); se += l[m]; }
  float inv = 1.0f / se, acc = 0.0f;
#pragma unroll
  for (int m = 0; m < 8; ++m)
    acc += l[m] * inv * xln[((size_t)(nt * 8 + m)) * HD + c];
  unsafeAtomicAdd(out + (size_t)batch[nt] * HD + c, acc);
}

// ---------------------------------------------------------------------------
// Host driver
// ---------------------------------------------------------------------------
extern "C" void kernel_launch(void* const* d_in, const int* in_sizes, int n_in,
                              void* d_out, int out_size, void* d_ws, size_t ws_size,
                              hipStream_t stream) {
  (void)n_in; (void)out_size; (void)ws_size;

  const int* x_ids   = (const int*)d_in[0];
  const int* ea_ids  = (const int*)d_in[1];
  const int* ei_src  = (const int*)d_in[2];
  const int* ei_dst  = ei_src + NE;
  const unsigned char* validm = (const unsigned char*)d_in[3];
  const float* log_probs = (const float*)d_in[4];
  const int*   batch     = (const int*)d_in[5];
  const float* atom_emb  = (const float*)d_in[6];
  const float* bond_emb  = (const float*)d_in[7];
  const float* rwse_W    = (const float*)d_in[8];
  const float* rwse_b    = (const float*)d_in[9];
  const float* logp_W    = (const float*)d_in[10];
  const float* logp_b    = (const float*)d_in[11];
  const float* lnf_g     = (const float*)d_in[12];
  const float* lnf_b     = (const float*)d_in[13];

  // nested param order: insertion ({eps,W1,b1,W2,b2}) vs alphabetical — detect
  const bool nsort = (in_sizes[14] != 1);
  int gEps[4], gW1[4], gB1[4], gW2[4], gB2[4];
  for (int l = 0; l < 4; ++l) {
    int b = 14 + l * 5;
    if (!nsort) { gEps[l]=b; gW1[l]=b+1; gB1[l]=b+2; gW2[l]=b+3; gB2[l]=b+4; }
    else        { gW1[l]=b; gW2[l]=b+1; gB1[l]=b+2; gB2[l]=b+3; gEps[l]=b+4; }
  }
  int tLn1g[4],tLn1b[4],tWq[4],tBq[4],tWk[4],tBk[4],tWv[4],tBv[4],
      tWo[4],tBo[4],tLn2g[4],tLn2b[4],tWf1[4],tBf1[4],tWf2[4],tBf2[4];
  for (int t = 0; t < 4; ++t) {
    int b = 34 + t * 16;
    if (!nsort) {
      tLn1g[t]=b+0; tLn1b[t]=b+1; tWq[t]=b+2; tBq[t]=b+3; tWk[t]=b+4; tBk[t]=b+5;
      tWv[t]=b+6; tBv[t]=b+7; tWo[t]=b+8; tBo[t]=b+9; tLn2g[t]=b+10; tLn2b[t]=b+11;
      tWf1[t]=b+12; tBf1[t]=b+13; tWf2[t]=b+14; tBf2[t]=b+15;
    } else { // sorted: Wf1,Wf2,Wk,Wo,Wq,Wv,bf1,bf2,bk,bo,bq,bv,ln1_b,ln1_g,ln2_b,ln2_g
      tWf1[t]=b+0; tWf2[t]=b+1; tWk[t]=b+2; tWo[t]=b+3; tWq[t]=b+4; tWv[t]=b+5;
      tBf1[t]=b+6; tBf2[t]=b+7; tBk[t]=b+8; tBo[t]=b+9; tBq[t]=b+10; tBv[t]=b+11;
      tLn1b[t]=b+12; tLn1g[t]=b+13; tLn2b[t]=b+14; tLn2g[t]=b+15;
    }
  }

  // ---- workspace layout (~405 MB) ----
  char* ws = (char*)d_ws;
  const size_t ABYTES = (size_t)NFN * HD * 4;            // 134,217,728
  float*    h_buf   = (float*)ws;
  char*     region1 = ws + ABYTES;                       // agg / transformer scratch
  char*     region2 = ws + 2 * ABYTES;                   // T+rw early; u16/t16 in GINE
  _Float16* w16base = (_Float16*)(ws + 3 * ABYTES);      // converted weights (~1.8MB)

  float*    agg  = (float*)region1;
  _Float16* u16  = (_Float16*)region2;
  _Float16* t16  = (_Float16*)(region2 + ABYTES / 2);
  float*    Tbuf = (float*)region2;                      // 16 MB, dead before u16 used
  float*    rwb  = (float*)(region2 + (size_t)NSUB * 256 * 4);

  float*    xbuf = (float*)region1;                      // transformer aliases (GINE done)
  _Float16* xn16 = (_Float16*)(region1 + ((size_t)9  << 20));
  float*    qbuf = (float*)(region1 + ((size_t)14 << 20));
  float*    kbuf = (float*)(region1 + ((size_t)23 << 20));
  float*    vbuf = (float*)(region1 + ((size_t)32 << 20));
  _Float16* o16  = (_Float16*)(region1 + ((size_t)41 << 20));
  _Float16* ff16 = (_Float16*)(region1 + ((size_t)46 << 20));
  float*    xln  = (float*)(region1 + ((size_t)64 << 20));

  _Float16* gW1t[4], *gW2t[4], *wQ[4], *wK[4], *wV[4], *wO[4], *wF1[4], *wF2[4];
  for (int l = 0; l < 4; ++l) {
    gW1t[l] = w16base + (size_t)(l * 2) * 16384;
    gW2t[l] = w16base + (size_t)(l * 2 + 1) * 16384;
  }
  for (int t = 0; t < 4; ++t) {
    _Float16* base = w16base + (size_t)8 * 16384 + (size_t)t * 196608;
    wQ[t] = base; wK[t] = base + 16384; wV[t] = base + 32768; wO[t] = base + 49152;
    wF1[t] = base + 65536; wF2[t] = base + 131072;
  }

  auto conv = [&](int idx, _Float16* dst, int Kd, int Nd) {
    int nel = Kd * Nd;
    k_convert_wt<<<(nel + 255) / 256, 256, 0, stream>>>((const float*)d_in[idx], dst, Kd, Nd);
  };

  // ---- stage 0: convert all weights to f16 transposed layout ----
  for (int l = 0; l < 4; ++l) { conv(gW1[l], gW1t[l], HD, HD); conv(gW2[l], gW2t[l], HD, HD); }
  for (int t = 0; t < 4; ++t) {
    conv(tWq[t], wQ[t], HD, HD);  conv(tWk[t], wK[t], HD, HD);
    conv(tWv[t], wV[t], HD, HD);  conv(tWo[t], wO[t], HD, HD);
    conv(tWf1[t], wF1[t], HD, FFD); conv(tWf2[t], wF2[t], FFD, HD);
  }

  // ---- stage 1: RWSE ----
  hipMemsetAsync(Tbuf, 0, (size_t)NSUB * 256 * 4, stream);
  k_build_T<<<NE / 256, 256, 0, stream>>>(ei_src, ei_dst, Tbuf);
  k_norm_T<<<(NSUB * 16) / 256, 256, 0, stream>>>(Tbuf);
  k_rwse<<<NSUB / 16, 256, 0, stream>>>(Tbuf, rwb);

  // ---- stage 2: init h ----
  k_init_h<<<((size_t)NFN * HD) / 256, 256, 0, stream>>>(
      x_ids, validm, log_probs, atom_emb, rwb, rwse_W, rwse_b, logp_W, logp_b, h_buf);

  // ---- stage 3: GINE x4 ----
  const int gemmH_blocks  = (NFN / 16) * (HD / 16) / 8;      // 16384
  for (int l = 0; l < 4; ++l) {
    hipMemsetAsync(agg, 0, ABYTES, stream);
    k_gine_scatter<<<(NE * 32) / 256, 256, 0, stream>>>(ei_src, ei_dst, ea_ids, bond_emb,
                                                        h_buf, agg);
    k_gine_fuse<<<((size_t)NFN * HD / 4) / 256, 256, 0, stream>>>(
        h_buf, agg, (const float*)d_in[gEps[l]], u16);
    // t = relu(u @ W1 + b1)  -> f16
    k_gemm_wmma<true, false, false, true><<<gemmH_blocks, 256, 0, stream>>>(
        u16, gW1t[l], (const float*)d_in[gB1[l]], nullptr, nullptr, nullptr, t16, HD, HD);
    // h = (h + t @ W2 + b2) * valid  -> f32 (in place)
    k_gemm_wmma<false, true, true, false><<<gemmH_blocks, 256, 0, stream>>>(
        t16, gW2t[l], (const float*)d_in[gB2[l]], h_buf, validm, h_buf, nullptr, HD, HD);
  }

  // ---- stage 4: gather roots into transformer state ----
  k_copy_roots<<<(NTOK * MTOK * HD) / 256, 256, 0, stream>>>(h_buf, xbuf);

  // ---- stage 5: transformer x4 ----
  const int rowsT = NTOK * MTOK;                              // 16384
  const int gQKV_blocks = (rowsT / 16) * (HD / 16) / 8;       // 1024
  const int gF1_blocks  = (rowsT / 16) * (FFD / 16) / 8;      // 4096
  for (int t = 0; t < 4; ++t) {
    k_ln<true><<<rowsT / 8, 256, 0, stream>>>(xbuf, (const float*)d_in[tLn1g[t]],
        (const float*)d_in[tLn1b[t]], nullptr, xn16, rowsT);
    k_gemm_wmma<false, false, false, false><<<gQKV_blocks, 256, 0, stream>>>(
        xn16, wQ[t], (const float*)d_in[tBq[t]], nullptr, nullptr, qbuf, nullptr, HD, HD);
    k_gemm_wmma<false, false, false, false><<<gQKV_blocks, 256, 0, stream>>>(
        xn16, wK[t], (const float*)d_in[tBk[t]], nullptr, nullptr, kbuf, nullptr, HD, HD);
    k_gemm_wmma<false, false, false, false><<<gQKV_blocks, 256, 0, stream>>>(
        xn16, wV[t], (const float*)d_in[tBv[t]], nullptr, nullptr, vbuf, nullptr, HD, HD);
    k_attn<<<(NTOK * 32) / 256, 256, 0, stream>>>(qbuf, kbuf, vbuf, log_probs, o16);
    // x = x + o @ Wo + bo
    k_gemm_wmma<false, true, false, false><<<gQKV_blocks, 256, 0, stream>>>(
        o16, wO[t], (const float*)d_in[tBo[t]], xbuf, nullptr, xbuf, nullptr, HD, HD);
    k_ln<true><<<rowsT / 8, 256, 0, stream>>>(xbuf, (const float*)d_in[tLn2g[t]],
        (const float*)d_in[tLn2b[t]], nullptr, xn16, rowsT);
    // f = relu(hN @ Wf1 + bf1) -> f16 (rows x 512)
    k_gemm_wmma<true, false, false, true><<<gF1_blocks, 256, 0, stream>>>(
        xn16, wF1[t], (const float*)d_in[tBf1[t]], nullptr, nullptr, nullptr, ff16, HD, FFD);
    // x = x + f @ Wf2 + bf2
    k_gemm_wmma<false, true, false, false><<<gQKV_blocks, 256, 0, stream>>>(
        ff16, wF2[t], (const float*)d_in[tBf2[t]], xbuf, nullptr, xbuf, nullptr, FFD, HD);
  }

  // ---- stage 6: final LN + softmax-weighted pool + segment sum ----
  k_ln<false><<<rowsT / 8, 256, 0, stream>>>(xbuf, lnf_g, lnf_b, xln, nullptr, rowsT);
  hipMemsetAsync(d_out, 0, (size_t)NB * HD * 4, stream);
  k_final<<<(NTOK * HD) / 256, 256, 0, stream>>>(xln, log_probs, batch, (float*)d_out);
}
